// MultiHeadAttention_10814727652024
// MI455X (gfx1250) — matrix operand order repaired and checked by reading, not compiled, not run on hardware
//
#include <hip/hip_runtime.h>
#include <hip/hip_bf16.h>

typedef __attribute__((ext_vector_type(16))) _Float16 v16h;
typedef __attribute__((ext_vector_type(8)))  float    v8f;
typedef __attribute__((ext_vector_type(4)))  unsigned int v4u;   // 16 bytes
typedef __attribute__((ext_vector_type(8)))  unsigned int v8u;   // 32 bytes

union FragH {
    v16h v;
    v4u  q[2];
    _Float16 h[16];
};

// A-matrix 16x32 f16 fragment (ISA 7.12.2): lane row = lane&15, khalf = lane>>4.
// VGPRs 0..3 hold K = khalf*8 + 0..7 ; VGPRs 4..7 hold K = 16 + khalf*8 + 0..7.
__device__ inline v16h ld_frag_a(const _Float16* base, int ldh, int lane) {
    FragH f;
    const _Float16* p = base + (lane & 15) * ldh + ((lane >> 4) << 3);
    f.q[0] = *(const v4u*)(p);
    f.q[1] = *(const v4u*)(p + 16);
    return f.v;
}

// B-matrix 32x16 f16 fragment: lane col = lane&15, khalf = lane>>4.
// Lane holds K = khalf*8 + 0..7 and K = 16 + khalf*8 + 0..7 (same as A). LDS tile stored [n][k].
__device__ inline v16h ld_frag_b(const _Float16* base, int ldh, int lane) {
    FragH f;
    const _Float16* p = base + (lane & 15) * ldh + ((lane >> 4) << 3);
    f.q[0] = *(const v4u*)(p);
    f.q[1] = *(const v4u*)(p + 16);
    return f.v;
}

// ---------------------------------------------------------------------------
// Tensor Data Mover: DMA a contiguous run of `nhalf` f16 elements from global
// memory into LDS at byte offset lds_addr (D# per CDNA5 ISA
// §8: 2D tensor, one row of nhalf elements, tile == tensor row).  Issued by a
// single wave; completion tracked with TENSORcnt.
// ---------------------------------------------------------------------------
__device__ inline void tdm_load_f16(unsigned lds_addr, const _Float16* gptr,
                                    unsigned nhalf) {
    unsigned long long ga = (unsigned long long)(uintptr_t)gptr;
    v4u g0;
    g0[0] = 1u;                                        // count=1, is_restore=0
    g0[1] = lds_addr;                                  // lds_addr (bytes)
    g0[2] = (unsigned)ga;                              // global_addr[31:0]
    g0[3] = (unsigned)((ga >> 32) & 0x1FFFFFFu)        // global_addr[56:32]
            | (2u << 30);                              // type=2 ("image")
    v8u g1;
    g1[0] = (1u << 16);                                // wg_mask=0, data_size=1 (2B)
    g1[1] = (nhalf & 0xFFFFu) << 16;                   // tensor_dim0[15:0]
    g1[2] = (nhalf >> 16) | (1u << 16);                // tensor_dim0[31:16] | tensor_dim1=1
    g1[3] = (nhalf & 0xFFFFu) << 16;                   // tensor_dim1 hi=0 | tile_dim0
    g1[4] = 1u;                                        // tile_dim1=1, tile_dim2=0
    g1[5] = nhalf;                                     // tensor_dim0_stride[31:0]
    g1[6] = 0u;                                        // stride0 hi | stride1 lo
    g1[7] = 0u;                                        // stride1 hi
    asm volatile("tensor_load_to_lds %0, %1" :: "s"(g0), "s"(g1) : "memory");
}

#define TM 128
#define TN 128
#define TK 32
#define DMODEL 1024
#define SEQ 2048
#define NB 4
#define NH 16
#define DKH 64
#define QTILE 128   // attention q rows per block (8 waves x 16 rows)

// ---------------------------------------------------------------------------
// Projection GEMM: C[m,n] = X[m,:] (f32) @ W[:,n] (f32) + bias[n], output f16
// written in head layout [B, H, S, DK].  M = B*S = 8192, N = K = 1024.
// ---------------------------------------------------------------------------
__global__ __launch_bounds__(256)
void proj_gemm_kernel(const float* __restrict__ X, const float* __restrict__ W,
                      const float* __restrict__ bias, _Float16* __restrict__ Out)
{
    __shared__ _Float16 As[TM * TK];   // [m][k]
    __shared__ _Float16 Bs[TN * TK];   // [n][k]  (transposed for B-fragment loads)

    const int tid  = threadIdx.x;
    const int lane = tid & 31;
    const int w    = tid >> 5;     // 0..7
    const int wm   = w & 1;        // 2 waves along M (64 rows each)
    const int wn   = w >> 1;       // 4 waves along N (32 cols each)
    const int m0   = blockIdx.x * TM;
    const int n0   = blockIdx.y * TN;

    v8f c[4][2] = {};

    for (int kb = 0; kb < DMODEL; kb += TK) {
        // uniform prefetch distances (0 on last iteration -> no branch, no OOB)
        const size_t pfA = (kb + TK < DMODEL) ? (size_t)TK : 0;
        const size_t pfB = (kb + TK < DMODEL) ? (size_t)TK * DMODEL : 0;

        // A tile: 128x32 f32 -> f16, 1024 float4 slots over 256 threads
#pragma unroll
        for (int it = 0; it < 4; ++it) {
            int s   = tid + it * 256;
            int row = s >> 3;
            int c4  = (s & 7) << 2;
            const float* src = X + (size_t)(m0 + row) * DMODEL + kb + c4;
            float4 fv = *(const float4*)src;
            __builtin_prefetch(src + pfA, 0, 1);
            _Float16* dst = &As[row * TK + c4];
            dst[0] = (_Float16)fv.x; dst[1] = (_Float16)fv.y;
            dst[2] = (_Float16)fv.z; dst[3] = (_Float16)fv.w;
        }
        // B tile: W[k][n] 32x128 f32 -> f16 transposed into Bs[n][k]
#pragma unroll
        for (int it = 0; it < 4; ++it) {
            int s  = tid + it * 256;
            int kk = s >> 5;
            int n4 = (s & 31) << 2;
            const float* src = W + (size_t)(kb + kk) * DMODEL + n0 + n4;
            float4 fv = *(const float4*)src;
            __builtin_prefetch(src + pfB, 0, 1);
            Bs[(n4 + 0) * TK + kk] = (_Float16)fv.x;
            Bs[(n4 + 1) * TK + kk] = (_Float16)fv.y;
            Bs[(n4 + 2) * TK + kk] = (_Float16)fv.z;
            Bs[(n4 + 3) * TK + kk] = (_Float16)fv.w;
        }
        __syncthreads();

        v16h a[4], b[2];
        for (int i = 0; i < 4; ++i) a[i] = ld_frag_a(&As[(wm * 64 + i * 16) * TK], TK, lane);
        for (int j = 0; j < 2; ++j) b[j] = ld_frag_b(&Bs[(wn * 32 + j * 16) * TK], TK, lane);
        for (int i = 0; i < 4; ++i)
            for (int j = 0; j < 2; ++j)
                c[i][j] = __builtin_amdgcn_wmma_f32_16x16x32_f16(
                    false, a[i], false, b[j], (short)0, c[i][j], false, false);
        __syncthreads();
    }

    // Epilogue: bias + f16 store in [B,H,S,DK] layout
    const int col = lane & 15, hh = lane >> 4;
    for (int i = 0; i < 4; ++i)
        for (int j = 0; j < 2; ++j) {
            int n    = n0 + wn * 32 + j * 16 + col;
            int hidx = n >> 6;
            int dk   = n & 63;
            float bval = bias[n];
            for (int v = 0; v < 8; ++v) {
                int m  = m0 + wm * 64 + i * 16 + v + (hh << 3);
                int bb = m >> 11;          // / SEQ
                int ss = m & (SEQ - 1);
                Out[((size_t)((bb * NH + hidx) * SEQ + ss) << 6) + dk] =
                    (_Float16)(c[i][j][v] + bval);
            }
        }
}

// ---------------------------------------------------------------------------
// Flash attention over one (batch*head, 128-row q tile).  8 waves x 16 q rows
// (128-row tile halves K/V re-read traffic vs 64 rows).  Q and K tiles are
// DMA'd into LDS by the Tensor Data Mover (wave 0 issues, TENSORcnt wait,
// workgroup barrier publishes).  V is loaded/transposed by all threads,
// overlapping with the in-flight TDM transfer.
// ---------------------------------------------------------------------------
__global__ __launch_bounds__(256)
void attn_kernel(const _Float16* __restrict__ Qh, const _Float16* __restrict__ Kh,
                 const _Float16* __restrict__ Vh, _Float16* __restrict__ Ao)
{
    __shared__ _Float16 Qs[QTILE * 64];   // [qrow][dk]
    __shared__ _Float16 Ks[64 * 64];      // [key][dk]  == Bt layout for Q*K^T
    __shared__ _Float16 Vt[64 * 64];      // [dk][key]  == Bt layout for P*V
    __shared__ _Float16 Ps[8][16 * 64];   // per-wave P tile [qrow][key]

    const int tid  = threadIdx.x;
    const int lane = tid & 31;
    const int w    = tid >> 5;            // 0..7
    const int bh   = blockIdx.x;          // 0..63
    const int b    = bh >> 4;
    const int h    = bh & 15;
    const int qt   = blockIdx.y;          // 0..15

    const _Float16* Qbase = Qh + ((size_t)bh * SEQ + qt * QTILE) * DKH;
    const _Float16* Kbase = Kh + (size_t)bh * SEQ * DKH;
    const _Float16* Vbase = Vh + (size_t)bh * SEQ * DKH;

    const unsigned QsAddr = (unsigned)(uintptr_t)&Qs[0];
    const unsigned KsAddr = (unsigned)(uintptr_t)&Ks[0];

    // Q tile via TDM (contiguous 128x64 f16 = 16KB)
    if (w == 0) {
        tdm_load_f16(QsAddr, Qbase, QTILE * DKH);
        __builtin_amdgcn_s_wait_tensorcnt(0);
    }
    __syncthreads();

    v16h aq[2];
    aq[0] = ld_frag_a(&Qs[(w * 16) * 64 + 0],  64, lane);
    aq[1] = ld_frag_a(&Qs[(w * 16) * 64 + 32], 64, lane);

    v8f acc[4] = {};
    float mrow[8], lrow[8];
    for (int v = 0; v < 8; ++v) { mrow[v] = -1e30f; lrow[v] = 0.f; }
    const float scale = 0.125f;           // 1/sqrt(64)

    for (int kt = 0; kt < SEQ / 64; ++kt) {
        __syncthreads();                  // all readers of Ks/Vt from prev iter done
        // K tile via TDM, issued by wave 0; overlaps with V transpose below
        if (w == 0)
            tdm_load_f16(KsAddr, Kbase + (size_t)kt * 64 * DKH, 64 * DKH);
        const _Float16* Vsrc = Vbase + (size_t)kt * 64 * DKH;
#pragma unroll
        for (int it = 0; it < 2; ++it) {
            int s   = tid + it * 256;
            int key = s >> 3;
            int c8  = (s & 7) << 3;
            FragH t;
            t.q[0] = *(const v4u*)(Vsrc + key * 64 + c8);
            for (int j = 0; j < 8; ++j) Vt[(c8 + j) * 64 + key] = t.h[j];
        }
        if (w == 0) __builtin_amdgcn_s_wait_tensorcnt(0);
        __syncthreads();

        // scores for this wave's 16 q rows x 64 keys
        v8f sf[4] = {};
        for (int j = 0; j < 4; ++j) {
            v16h bk0 = ld_frag_b(&Ks[(j * 16) * 64 + 0],  64, lane);
            v16h bk1 = ld_frag_b(&Ks[(j * 16) * 64 + 32], 64, lane);
            sf[j] = __builtin_amdgcn_wmma_f32_16x16x32_f16(false, aq[0], false, bk0,
                                                           (short)0, sf[j], false, false);
            sf[j] = __builtin_amdgcn_wmma_f32_16x16x32_f16(false, aq[1], false, bk1,
                                                           (short)0, sf[j], false, false);
        }

        // online softmax: row v lives in 16 lanes of one half-wave (rows v+8*hh)
        const int prow = ((lane >> 4) << 3);
        const int pcol = lane & 15;
        for (int v = 0; v < 8; ++v) {
            float s0 = sf[0][v] * scale, s1 = sf[1][v] * scale;
            float s2 = sf[2][v] * scale, s3 = sf[3][v] * scale;
            float rm = fmaxf(fmaxf(s0, s1), fmaxf(s2, s3));
            for (int m = 1; m < 16; m <<= 1) rm = fmaxf(rm, __shfl_xor(rm, m, 32));
            float mnew = fmaxf(mrow[v], rm);
            float p0 = __expf(s0 - mnew), p1 = __expf(s1 - mnew);
            float p2 = __expf(s2 - mnew), p3 = __expf(s3 - mnew);
            float rs = p0 + p1 + p2 + p3;
            for (int m = 1; m < 16; m <<= 1) rs += __shfl_xor(rs, m, 32);
            float corr = __expf(mrow[v] - mnew);
            mrow[v] = mnew;
            lrow[v] = lrow[v] * corr + rs;
            for (int j = 0; j < 4; ++j) acc[j][v] *= corr;
            _Float16* pr = &Ps[w][(v + prow) * 64 + pcol];
            pr[0]  = (_Float16)p0;
            pr[16] = (_Float16)p1;
            pr[32] = (_Float16)p2;
            pr[48] = (_Float16)p3;
        }

        // acc += P @ V   (A = Ps[w] 16x64, B = Vt)
        for (int kk = 0; kk < 2; ++kk) {
            v16h ap = ld_frag_a(&Ps[w][kk * 32], 64, lane);
            for (int j = 0; j < 4; ++j) {
                v16h bv = ld_frag_b(&Vt[(j * 16) * 64 + kk * 32], 64, lane);
                acc[j] = __builtin_amdgcn_wmma_f32_16x16x32_f16(false, ap, false, bv,
                                                                (short)0, acc[j], false, false);
            }
        }
    }

    // Normalize and store merged-head f16 output [B,S,D]
    const int col = lane & 15, hh = lane >> 4;
    for (int j = 0; j < 4; ++j)
        for (int v = 0; v < 8; ++v) {
            int srow = qt * QTILE + w * 16 + v + (hh << 3);
            float o  = acc[j][v] / lrow[v];
            Ao[(size_t)(b * SEQ + srow) * DMODEL + h * DKH + j * 16 + col] = (_Float16)o;
        }
}

// ---------------------------------------------------------------------------
// Output GEMM: out[m,n] (f32) = Ah[m,:] (f16) @ Wo[:,n] (f32->f16) + bo[n]
// ---------------------------------------------------------------------------
__global__ __launch_bounds__(256)
void out_gemm_kernel(const _Float16* __restrict__ Xh, const float* __restrict__ W,
                     const float* __restrict__ bias, float* __restrict__ Out)
{
    __shared__ _Float16 As[TM * TK];
    __shared__ _Float16 Bs[TN * TK];

    const int tid  = threadIdx.x;
    const int lane = tid & 31;
    const int w    = tid >> 5;
    const int wm   = w & 1;
    const int wn   = w >> 1;
    const int m0   = blockIdx.x * TM;
    const int n0   = blockIdx.y * TN;

    v8f c[4][2] = {};

    for (int kb = 0; kb < DMODEL; kb += TK) {
        const size_t pfA = (kb + TK < DMODEL) ? (size_t)TK : 0;
        const size_t pfB = (kb + TK < DMODEL) ? (size_t)TK * DMODEL : 0;

#pragma unroll
        for (int it = 0; it < 2; ++it) {   // 512 16-byte chunks over 256 threads
            int s   = tid + it * 256;
            int row = s >> 2;
            int c8  = (s & 3) << 3;
            const _Float16* src = Xh + (size_t)(m0 + row) * DMODEL + kb + c8;
            *(v4u*)&As[row * TK + c8] = *(const v4u*)src;
            __builtin_prefetch(src + pfA, 0, 1);
        }
#pragma unroll
        for (int it = 0; it < 4; ++it) {
            int s  = tid + it * 256;
            int kk = s >> 5;
            int n4 = (s & 31) << 2;
            const float* src = W + (size_t)(kb + kk) * DMODEL + n0 + n4;
            float4 fv = *(const float4*)src;
            __builtin_prefetch(src + pfB, 0, 1);
            Bs[(n4 + 0) * TK + kk] = (_Float16)fv.x;
            Bs[(n4 + 1) * TK + kk] = (_Float16)fv.y;
            Bs[(n4 + 2) * TK + kk] = (_Float16)fv.z;
            Bs[(n4 + 3) * TK + kk] = (_Float16)fv.w;
        }
        __syncthreads();

        v16h a[4], b[2];
        for (int i = 0; i < 4; ++i) a[i] = ld_frag_a(&As[(wm * 64 + i * 16) * TK], TK, lane);
        for (int j = 0; j < 2; ++j) b[j] = ld_frag_b(&Bs[(wn * 32 + j * 16) * TK], TK, lane);
        for (int i = 0; i < 4; ++i)
            for (int j = 0; j < 2; ++j)
                c[i][j] = __builtin_amdgcn_wmma_f32_16x16x32_f16(
                    false, a[i], false, b[j], (short)0, c[i][j], false, false);
        __syncthreads();
    }

    const int col = lane & 15, hh = lane >> 4;
    for (int i = 0; i < 4; ++i)
        for (int j = 0; j < 2; ++j) {
            int n = n0 + wn * 32 + j * 16 + col;
            float bval = bias[n];
            for (int v = 0; v < 8; ++v) {
                int m = m0 + wm * 64 + i * 16 + v + (hh << 3);
                Out[(size_t)m * DMODEL + n] = c[i][j][v] + bval;
            }
        }
}

extern "C" void kernel_launch(void* const* d_in, const int* in_sizes, int n_in,
                              void* d_out, int out_size, void* d_ws, size_t ws_size,
                              hipStream_t stream) {
    (void)in_sizes; (void)n_in; (void)out_size; (void)ws_size;
    const float* q  = (const float*)d_in[0];
    const float* k  = (const float*)d_in[1];
    const float* v  = (const float*)d_in[2];
    const float* Wq = (const float*)d_in[3];
    const float* bq = (const float*)d_in[4];
    const float* Wk = (const float*)d_in[5];
    const float* bk = (const float*)d_in[6];
    const float* Wv = (const float*)d_in[7];
    const float* bv = (const float*)d_in[8];
    const float* Wo = (const float*)d_in[9];
    const float* bo = (const float*)d_in[10];
    float* out = (float*)d_out;

    _Float16* ws = (_Float16*)d_ws;
    const size_t HBUF = (size_t)NB * NH * SEQ * DKH;   // 8M halves = 16MB
    _Float16* Qh = ws;
    _Float16* Kh = ws + HBUF;
    _Float16* Vh = ws + 2 * HBUF;
    _Float16* Ao = ws + 3 * HBUF;

    dim3 gg(NB * SEQ / TM, DMODEL / TN);   // (64, 8)
    proj_gemm_kernel<<<gg, 256, 0, stream>>>(q, Wq, bq, Qh);
    proj_gemm_kernel<<<gg, 256, 0, stream>>>(k, Wk, bk, Kh);
    proj_gemm_kernel<<<gg, 256, 0, stream>>>(v, Wv, bv, Vh);
    attn_kernel<<<dim3(NB * NH, SEQ / QTILE), 256, 0, stream>>>(Qh, Kh, Vh, Ao);
    out_gemm_kernel<<<gg, 256, 0, stream>>>(Ao, Wo, bo, out);
}
